// CoAtNetRelativeAttention_2156073583035
// MI455X (gfx1250) — matrix operand order repaired and checked by reading, not compiled, not run on hardware
//
#include <hip/hip_runtime.h>

typedef _Float16 v16h __attribute__((ext_vector_type(16)));
typedef _Float16 v8h  __attribute__((ext_vector_type(8)));
typedef float    v8f  __attribute__((ext_vector_type(8)));

#define HEADS    8
#define HEAD_DIM 32
#define NTOK     1024   // 32*32 tokens
#define CCH      256    // channels
#define DIM      256    // HEADS*HEAD_DIM
#define NBIAS    3969   // (2*32-1)^2
#define KCH      32     // keys per streamed chunk
#define NCHUNK   (NTOK / KCH)
#define QK_SCALE 0.17677669529663687f  // 32^-0.5

// Build a 16-half fragment from two contiguous 16B segments.
__device__ __forceinline__ v16h ldfrag(const _Float16* p0, const _Float16* p1) {
  v8h a = *(const v8h*)p0;
  v8h b = *(const v8h*)p1;
  v16h r;
#pragma unroll
  for (int i = 0; i < 8; ++i) { r[i] = a[i]; r[i + 8] = b[i]; }
  return r;
}

__device__ __forceinline__ v8f wmma_f16(v16h a, v16h b, v8f c) {
  // D = A(16x32 f16) * B(32x16 f16) + C(16x16 f32)
  return __builtin_amdgcn_wmma_f32_16x16x32_f16(false, a, false, b, (short)0, c,
                                                false, false);
}

// ---------------------------------------------------------------------------
// Kernel 1: fused Q/K/V projection. One wave computes a 16-token tile x all
// 256 output features of one of {Q,K,V}. x is (b,c,n) channel-major f32.
//   Qf[( (b*8+h)*1024 + n )*32 + d]   (f16, d contiguous)
//   Kf same layout, pre-scaled by d^-1/2
//   Vt[( (b*8+h)*32 + d )*1024 + n]   (f16, transposed: n contiguous)
// ---------------------------------------------------------------------------
__global__ __launch_bounds__(32)
void qkv_kernel(const float* __restrict__ x, const float* __restrict__ Wq,
                const float* __restrict__ Wk, const float* __restrict__ Wv,
                _Float16* __restrict__ Qf, _Float16* __restrict__ Kf,
                _Float16* __restrict__ Vt) {
  const int lane = threadIdx.x;
  const int l16 = lane & 15;
  const int hi  = lane >> 4;
  const int nt  = blockIdx.x;   // token tile 0..63
  const int mat = blockIdx.y;   // 0=Q 1=K 2=V
  const int b   = blockIdx.z;
  const int n0  = nt * 16;
  const int xrow = n0 + l16;    // A-layout: lane%16 == row M
  const float* W = (mat == 0) ? Wq : (mat == 1) ? Wk : Wv;

  // Cache all 8 A-fragments of x^T (16 tokens x 32-channel chunks).
  v16h afr[8];
#pragma unroll
  for (int kc = 0; kc < 8; ++kc) {
    const int c0 = kc * 32;
    const int s0 = c0 + (hi ? 16 : 0);   // A lane-lo: K 0-7,  lane-hi: K 16-23
    const int s1 = c0 + (hi ? 24 : 8);   //            K 8-15,           24-31
    v16h a;
#pragma unroll
    for (int j = 0; j < 8; ++j) {
      a[j]     = (_Float16)x[((size_t)b * CCH + s0 + j) * NTOK + xrow];
      a[j + 8] = (_Float16)x[((size_t)b * CCH + s1 + j) * NTOK + xrow];
    }
    afr[kc] = a;
  }

#pragma unroll 1
  for (int et = 0; et < 16; ++et) {
    const int e = et * 16 + l16;  // feature column this lane owns (N)
    v8f acc = {};
#pragma unroll
    for (int kc = 0; kc < 8; ++kc) {
      const int c0 = kc * 32 + (hi ? 16 : 0);  // B lane-lo K 0-15, hi K 16-31
      const float* wrow = W + (size_t)e * CCH + c0;
      v16h bfr;
#pragma unroll
      for (int i = 0; i < 16; ++i) bfr[i] = (_Float16)wrow[i];
      acc = wmma_f16(afr[kc], bfr, acc);
    }
    const int h = e >> 5;
    const int d = e & 31;
    if (mat == 2) {
      _Float16* dst = Vt + ((size_t)(b * HEADS + h) * HEAD_DIM + d) * NTOK + n0;
#pragma unroll
      for (int r = 0; r < 8; ++r) dst[r + 8 * hi] = (_Float16)acc[r];
    } else {
      _Float16* dst = ((mat == 0) ? Qf : Kf) +
                      ((size_t)(b * HEADS + h) * NTOK + n0) * HEAD_DIM + d;
      const float s = (mat == 1) ? QK_SCALE : 1.0f;
#pragma unroll
      for (int r = 0; r < 8; ++r)
        dst[(size_t)(r + 8 * hi) * HEAD_DIM] = (_Float16)(acc[r] * s);
    }
  }
}

// ---------------------------------------------------------------------------
// Kernel 2: flash attention, 4 waves per block sharing async-staged K/V.
// Block = (b, h, group of 4 query tiles). Each 32-key chunk (2KB K + 2KB V)
// is DMA'd to LDS with global_load_async_to_lds_b128, double buffered so the
// next chunk's DMA overlaps WMMA on the current one.
// Output: Of[(b*1024+n)*256 + h*32 + d]  (f16, features contiguous)
// ---------------------------------------------------------------------------
__global__ __launch_bounds__(128)
void attn_kernel(const _Float16* __restrict__ Qf, const _Float16* __restrict__ Kf,
                 const _Float16* __restrict__ Vt, const float* __restrict__ rel_bias,
                 const int* __restrict__ rel_idx, _Float16* __restrict__ Of) {
  __shared__ __align__(16) _Float16 kbuf[2][KCH * HEAD_DIM];  // [key_local][d]
  __shared__ __align__(16) _Float16 vbuf[2][HEAD_DIM * KCH];  // [d][m_local]
  __shared__ __align__(16) _Float16 pbuf[4][16 * 32];         // per-wave P tile

  const int tid  = threadIdx.x;
  const int wave = tid >> 5;
  const int lane = tid & 31;
  const int l16 = lane & 15, hi = lane >> 4;
  const int it = blockIdx.x * 4 + wave;
  const int h = blockIdx.y, b = blockIdx.z;
  const int n0 = it * 16;
  const size_t bh = (size_t)(b * HEADS + h);

  // Async staging bases (GVS mode: SGPR 64-bit base + 32-bit VGPR offset).
  const _Float16* kcb = Kf + bh * NTOK * HEAD_DIM;  // + (j0+t)*32 halves, contiguous 2KB/chunk
  const _Float16* vcb = Vt + bh * HEAD_DIM * NTOK;  // + d*1024 + m0
  const int vd  = tid >> 2;  // 0..31 head-dim row
  const int vsg = tid & 3;   // 16B segment within a 64B row piece

  auto issue_chunk = [&](int jc) {
    const int bufi = jc & 1;
    const int j0 = jc * KCH;
    unsigned kl = (unsigned)(uintptr_t)(&kbuf[bufi][0]) + (unsigned)tid * 16u;
    unsigned kg = (unsigned)(j0 * HEAD_DIM * 2) + (unsigned)tid * 16u;
    asm volatile("global_load_async_to_lds_b128 %0, %1, %2"
                 :: "v"(kl), "v"(kg), "s"(kcb) : "memory");
    unsigned vl = (unsigned)(uintptr_t)(&vbuf[bufi][0]) + (unsigned)tid * 16u;
    unsigned vg = (unsigned)((vd * NTOK + j0) * 2) + (unsigned)vsg * 16u;
    asm volatile("global_load_async_to_lds_b128 %0, %1, %2"
                 :: "v"(vl), "v"(vg), "s"(vcb) : "memory");
  };

  issue_chunk(0);

  // Q fragment (A layout) straight from global.
  const _Float16* qbase = Qf + (bh * NTOK + n0 + l16) * HEAD_DIM;
  const v16h qa = ldfrag(qbase + (hi ? 16 : 0), qbase + (hi ? 24 : 8));

  const float* biasH = rel_bias + (size_t)h * NBIAS;
  int ibase[8];
#pragma unroll
  for (int r = 0; r < 8; ++r) ibase[r] = (n0 + r + 8 * hi) * NTOK;

  v8f acc0 = {}, acc1 = {};
  float mrow[8], lrow[8];
#pragma unroll
  for (int r = 0; r < 8; ++r) { mrow[r] = -3.0e38f; lrow[r] = 0.0f; }

  for (int jc = 0; jc < NCHUNK; ++jc) {
    if (jc + 1 < NCHUNK) {
      issue_chunk(jc + 1);
      asm volatile("s_wait_asynccnt 0x2" ::: "memory");  // chunk jc landed
    } else {
      asm volatile("s_wait_asynccnt 0x0" ::: "memory");
    }
    __syncthreads();  // publish LDS chunk across the 4 waves

    const _Float16* kb = kbuf[jc & 1];
    const _Float16* vb = vbuf[jc & 1];
    const int j0 = jc * KCH;
    const int kk = hi ? 16 : 0;

    // K fragments (B layout) for key tiles j0..+15 and j0+16..+31.
    const _Float16* kp0 = kb + l16 * HEAD_DIM + kk;
    const _Float16* kp1 = kb + (16 + l16) * HEAD_DIM + kk;
    const v16h kf0 = ldfrag(kp0, kp0 + 8);
    const v16h kf1 = ldfrag(kp1, kp1 + 8);
    v8f z = {};
    v8f s0 = wmma_f16(qa, kf0, z);
    v8f s1 = wmma_f16(qa, kf1, z);
    // relative position bias gather
#pragma unroll
    for (int r = 0; r < 8; ++r) {
      s0[r] += biasH[rel_idx[ibase[r] + j0 + l16]];
      s1[r] += biasH[rel_idx[ibase[r] + j0 + 16 + l16]];
    }
    // online softmax (row = 16-lane half-wave group)
    float pr0[8], pr1[8];
#pragma unroll
    for (int r = 0; r < 8; ++r) {
      float mv = fmaxf(s0[r], s1[r]);
#pragma unroll
      for (int off = 1; off < 16; off <<= 1)
        mv = fmaxf(mv, __shfl_xor(mv, off, 16));
      const float mnew = fmaxf(mrow[r], mv);
      const float corr = __expf(mrow[r] - mnew);
      mrow[r] = mnew;
      pr0[r] = __expf(s0[r] - mnew);
      pr1[r] = __expf(s1[r] - mnew);
      float sum = pr0[r] + pr1[r];
#pragma unroll
      for (int off = 1; off < 16; off <<= 1)
        sum += __shfl_xor(sum, off, 16);
      lrow[r] = lrow[r] * corr + sum;
      acc0[r] *= corr;
      acc1[r] *= corr;
    }
    // C-layout -> A-layout transpose of P through this wave's LDS slab
    _Float16* pw = pbuf[wave];
#pragma unroll
    for (int r = 0; r < 8; ++r) {
      pw[(r + 8 * hi) * 32 + l16]      = (_Float16)pr0[r];
      pw[(r + 8 * hi) * 32 + 16 + l16] = (_Float16)pr1[r];
    }
    asm volatile("s_wait_dscnt 0" ::: "memory");
    const _Float16* pb = pw + l16 * 32;
    const v16h pa = ldfrag(pb + (hi ? 16 : 0), pb + (hi ? 24 : 8));
    // V fragments (B layout, columns = head-dim) from LDS
    const _Float16* vp0 = vb + l16 * KCH + kk;
    const _Float16* vp1 = vb + (16 + l16) * KCH + kk;
    const v16h vf0 = ldfrag(vp0, vp0 + 8);
    const v16h vf1 = ldfrag(vp1, vp1 + 8);
    acc0 = wmma_f16(pa, vf0, acc0);
    acc1 = wmma_f16(pa, vf1, acc1);

    __syncthreads();  // all waves done with buf[jc&1] before it is re-filled
  }

  // normalize and store
#pragma unroll
  for (int r = 0; r < 8; ++r) {
    const float rcp = 1.0f / lrow[r];
    _Float16* dst = Of + ((size_t)b * NTOK + n0 + r + 8 * hi) * DIM + h * HEAD_DIM;
    dst[l16]      = (_Float16)(acc0[r] * rcp);
    dst[16 + l16] = (_Float16)(acc1[r] * rcp);
  }
}

// ---------------------------------------------------------------------------
// Kernel 3: output projection out = O @ Wo^T + bo, stored as (b, c, n) f32.
// One wave = one 16x16 (token x channel) output tile; K-loop over dim=256.
// ---------------------------------------------------------------------------
__global__ __launch_bounds__(32)
void proj_kernel(const _Float16* __restrict__ Of, const float* __restrict__ Wo,
                 const float* __restrict__ bo, float* __restrict__ out) {
  const int lane = threadIdx.x;
  const int l16 = lane & 15, hi = lane >> 4;
  const int nt = blockIdx.x, ct = blockIdx.y, b = blockIdx.z;
  const int n0 = nt * 16;
  const int c  = ct * 16 + l16;  // channel column this lane owns (N)
  v8f acc = {};
  const _Float16* abase = Of + ((size_t)b * NTOK + n0 + l16) * DIM;
#pragma unroll
  for (int kc = 0; kc < 8; ++kc) {
    const int e0 = kc * 32;
    const v16h afr = ldfrag(abase + e0 + (hi ? 16 : 0), abase + e0 + (hi ? 24 : 8));
    const float* wrow = Wo + (size_t)c * DIM + e0 + (hi ? 16 : 0);
    v16h bfr;
#pragma unroll
    for (int i = 0; i < 16; ++i) bfr[i] = (_Float16)wrow[i];
    acc = wmma_f16(afr, bfr, acc);
  }
  const float bc = bo[c];
  float* dst = out + ((size_t)b * CCH + c) * NTOK + n0;
#pragma unroll
  for (int r = 0; r < 8; ++r) dst[r + 8 * hi] = acc[r] + bc;
}

extern "C" void kernel_launch(void* const* d_in, const int* in_sizes, int n_in,
                              void* d_out, int out_size, void* d_ws, size_t ws_size,
                              hipStream_t stream) {
  const float* x  = (const float*)d_in[0];
  const float* Wq = (const float*)d_in[1];
  const float* Wk = (const float*)d_in[2];
  const float* Wv = (const float*)d_in[3];
  const float* Wo = (const float*)d_in[4];
  const float* bo = (const float*)d_in[5];
  const float* rb = (const float*)d_in[6];
  const int*   ri = (const int*)d_in[7];
  float* out = (float*)d_out;

  const int B = in_sizes[0] / (CCH * NTOK);  // 16
  const size_t perMat = (size_t)B * HEADS * NTOK * HEAD_DIM;  // f16 elems
  _Float16* Qf = (_Float16*)d_ws;
  _Float16* Kf = Qf + perMat;
  _Float16* Vt = Kf + perMat;
  _Float16* Of = Vt + perMat;

  qkv_kernel <<<dim3(NTOK / 16, 3, B),          32, 0, stream>>>(x, Wq, Wk, Wv, Qf, Kf, Vt);
  attn_kernel<<<dim3(NTOK / 64, HEADS, B),     128, 0, stream>>>(Qf, Kf, Vt, rb, ri, Of);
  proj_kernel<<<dim3(NTOK / 16, CCH / 16, B),   32, 0, stream>>>(Of, Wo, bo, out);
}
